// InteractionBlock_47382079210051
// MI455X (gfx1250) — matrix operand order repaired and checked by reading, not compiled, not run on hardware
//
#include <hip/hip_runtime.h>
#include <math.h>

#define N_NODES 20000
#define N_EDGES 320000
#define NF      256
#define N_GAUSS 50
#define LOG2F_  0.6931471805599453f
#define PI_OVER_CUT 0.31415926535897931f  /* pi / 10 */

typedef __attribute__((ext_vector_type(16))) _Float16 v16h;
typedef __attribute__((ext_vector_type(8)))  float    v8f;

enum { EPI_STORE_F32 = 0, EPI_STORE_F16 = 1, EPI_SCATTER = 2 };

__device__ __forceinline__ float ssp_f(float v) {
    // softplus(v) - log(2), numerically stable
    float sp = (v > 20.f) ? v : log1pf(__expf(v));
    return sp - LOG2F_;
}

// D[M,256] = epilogue(A[M,K(lda)] @ B[K,256]); block = 32 rows x 256 cols,
// 4 wave32s, each wave computes a 16x128 slab as 8 x v_wmma_f32_16x16x32_f16.
// EPI_SCATTER fuses the CFConv message+scatter: rows are edges e, and the
// epilogue does agg[dst[e]] += (acc+b)*C[e] * H[src[e]] with hw fp32 atomics,
// so the [E,256] filter matrix W never touches HBM.
template<typename AT, bool HAS_BIAS, bool DO_SSP, int EPI>
__global__ __launch_bounds__(128)
void gemm_wmma(const AT* A, int lda, int K,
               const float* __restrict__ B,
               const float* __restrict__ bias,
               void* Dst,
               const int* __restrict__ ei,     // scatter only: [2,E]
               const float* __restrict__ Cenv, // scatter only: [E]
               const float* __restrict__ Hgat) // scatter only: [N,256]
{
    __shared__ alignas(32) _Float16 As[32][40];    // [row][k] tile, padded
    __shared__ alignas(32) _Float16 Bst[256][48];  // B transposed [n][k], padded

    const int tid  = threadIdx.x;
    const int lane = tid & 31;
    const int wave = tid >> 5;
    const int row0 = blockIdx.x * 32;
    const int rowSlab = (wave >> 1) * 16;   // 0 or 16
    const int colHalf = (wave & 1) * 128;   // 0 or 128

    const bool vecA = ((lda & 3) == 0) && ((K & 31) == 0);

    v8f acc[8] = {};

    for (int k0 = 0; k0 < K; k0 += 32) {
        if (k0 + 32 < K) __builtin_prefetch(B + (size_t)(k0 + 32) * NF, 0, 0);

        // ---- Stage B^T tile (32 k x 256 n), fp32 -> f16, float4 loads ----
        for (int idx = tid; idx < 32 * 64; idx += 128) {
            const int kk = idx >> 6, n4 = (idx & 63) << 2;
            float4 v = make_float4(0.f, 0.f, 0.f, 0.f);
            if (k0 + kk < K) v = *(const float4*)(B + (size_t)(k0 + kk) * NF + n4);
            Bst[n4 + 0][kk] = (_Float16)v.x;
            Bst[n4 + 1][kk] = (_Float16)v.y;
            Bst[n4 + 2][kk] = (_Float16)v.z;
            Bst[n4 + 3][kk] = (_Float16)v.w;
        }

        // ---- Stage A tile (32 rows x 32 k) ----
        if constexpr (sizeof(AT) == 2) {
            // f16 source: straight 16B copies (K is a multiple of 32 here)
            for (int idx = tid; idx < 32 * 4; idx += 128) {
                const int rr = idx >> 2, c8 = (idx & 3) << 3;
                float4 raw = *(const float4*)((const _Float16*)A +
                                              (size_t)(row0 + rr) * lda + k0 + c8);
                *(float4*)&As[rr][c8] = raw;
            }
        } else {
            if (vecA) {
                for (int idx = tid; idx < 32 * 8; idx += 128) {
                    const int rr = idx >> 3, c4 = (idx & 7) << 2;
                    float4 v = *(const float4*)((const float*)A +
                                                (size_t)(row0 + rr) * lda + k0 + c4);
                    As[rr][c4 + 0] = (_Float16)v.x;
                    As[rr][c4 + 1] = (_Float16)v.y;
                    As[rr][c4 + 2] = (_Float16)v.z;
                    As[rr][c4 + 3] = (_Float16)v.w;
                }
            } else {  // unaligned / ragged K (edge_attr, K=50)
                for (int idx = tid; idx < 32 * 32; idx += 128) {
                    const int rr = idx >> 5, kk = idx & 31;
                    const int gk = k0 + kk;
                    float v = (gk < K) ? ((const float*)A)[(size_t)(row0 + rr) * lda + gk] : 0.f;
                    As[rr][kk] = (_Float16)v;
                }
            }
        }
        __syncthreads();

        // ---- A fragment: ISA 16-bit A 16x32 lane layout ----
        const int m  = rowSlab + (lane & 15);
        const int kb = (lane >> 4) << 3;   // 0 | 8
        v16h afrag;
        #pragma unroll
        for (int i = 0; i < 8; ++i) {
            afrag[i]     = As[m][kb + i];
            afrag[i + 8] = As[m][kb + 16 + i];
        }
        const int kbB = (lane >> 4) << 3;  // 0 | 8
        #pragma unroll
        for (int t = 0; t < 8; ++t) {
            const int nn = colHalf + t * 16 + (lane & 15);
            v16h bfrag;
            #pragma unroll
            for (int i = 0; i < 16; ++i) bfrag[i] = Bst[nn][kbB + (i & 7) + ((i >> 3) << 4)];
            acc[t] = __builtin_amdgcn_wmma_f32_16x16x32_f16(
                false, afrag, false, bfrag, (short)0, acc[t], false, false);
        }
        __syncthreads();
    }

    // ---- Epilogue: C/D layout -> lane holds (M = r + 8*(lane>=16), N = lane&15)
    const int mrow = row0 + rowSlab + ((lane >> 4) << 3);
    const int ncol = colHalf + (lane & 15);

    if constexpr (EPI == EPI_SCATTER) {
        float bv[8];
        #pragma unroll
        for (int t = 0; t < 8; ++t) bv[t] = bias[ncol + t * 16];
        float* agg = (float*)Dst;
        #pragma unroll
        for (int r = 0; r < 8; ++r) {
            const int e    = mrow + r;           // edge id
            const int srcn = ei[e];
            const int dstn = ei[N_EDGES + e];
            const float ce = Cenv[e];
            const float* hrow = Hgat + (size_t)srcn * NF;
            float*       arow = agg  + (size_t)dstn * NF;
            #pragma unroll
            for (int t = 0; t < 8; ++t) {
                const int col = ncol + t * 16;
                const float w = (acc[t][r] + bv[t]) * ce;
                unsafeAtomicAdd(arow + col, w * hrow[col]);
            }
        }
    } else {
        #pragma unroll
        for (int t = 0; t < 8; ++t) {
            const int col = ncol + t * 16;
            const float bvv = HAS_BIAS ? bias[col] : 0.f;
            #pragma unroll
            for (int r = 0; r < 8; ++r) {
                float v = acc[t][r] + bvv;
                if (DO_SSP) v = ssp_f(v);
                if constexpr (EPI == EPI_STORE_F16)
                    ((_Float16*)Dst)[(size_t)(mrow + r) * NF + col] = (_Float16)v;
                else
                    ((float*)Dst)[(size_t)(mrow + r) * NF + col] = v;
            }
        }
    }
}

__global__ __launch_bounds__(256)
void envelope_kernel(const float* __restrict__ ew, float* __restrict__ C, int E) {
    int i = blockIdx.x * blockDim.x + threadIdx.x;
    if (i < E) C[i] = 0.5f * (__cosf(ew[i] * PI_OVER_CUT) + 1.f);
}

__global__ __launch_bounds__(256)
void zero_kernel(float* __restrict__ p, long n) {
    long i = blockIdx.x * (long)blockDim.x + threadIdx.x;
    if (i < n) p[i] = 0.f;
}

static inline size_t align256(size_t b) { return (b + 255) & ~(size_t)255; }

extern "C" void kernel_launch(void* const* d_in, const int* in_sizes, int n_in,
                              void* d_out, int out_size, void* d_ws, size_t ws_size,
                              hipStream_t stream)
{
    const float* x     = (const float*)d_in[0];
    const int*   ei    = (const int*)  d_in[1];
    const float* ew    = (const float*)d_in[2];
    const float* eattr = (const float*)d_in[3];
    const float* w1    = (const float*)d_in[4];
    const float* b1    = (const float*)d_in[5];
    const float* w2    = (const float*)d_in[6];
    const float* b2    = (const float*)d_in[7];
    const float* lin1  = (const float*)d_in[8];
    const float* lin2  = (const float*)d_in[9];
    const float* lin2b = (const float*)d_in[10];
    const float* linw  = (const float*)d_in[11];
    const float* linb  = (const float*)d_in[12];
    float* out = (float*)d_out;

    // Workspace layout (~206 MB)
    char* ws = (char*)d_ws;
    size_t off = 0;
    float*    Cbuf = (float*)   (ws + off); off += align256((size_t)N_EDGES * 4);
    _Float16* HID  = (_Float16*)(ws + off); off += align256((size_t)N_EDGES * NF * 2);
    float*    Hbuf = (float*)   (ws + off); off += align256((size_t)N_NODES * NF * 4);
    float*    AGG  = (float*)   (ws + off); off += align256((size_t)N_NODES * NF * 4);

    // 1) cosine cutoff envelope C[e]
    envelope_kernel<<<(N_EDGES + 255) / 256, 256, 0, stream>>>(ew, Cbuf, N_EDGES);

    // 2) zero the aggregation buffer
    zero_kernel<<<(int)(((long)N_NODES * NF + 255) / 256), 256, 0, stream>>>(
        AGG, (long)N_NODES * NF);

    // 3) HID = ssp(edge_attr @ w1 + b1), stored f16       [E,256]
    gemm_wmma<float, true, true, EPI_STORE_F16><<<N_EDGES / 32, 128, 0, stream>>>(
        eattr, N_GAUSS, N_GAUSS, w1, b1, HID, nullptr, nullptr, nullptr);

    // 4) H = x @ lin1                                     [N,256]
    gemm_wmma<float, false, false, EPI_STORE_F32><<<N_NODES / 32, 128, 0, stream>>>(
        x, NF, NF, lin1, nullptr, Hbuf, nullptr, nullptr, nullptr);

    // 5) fused: W = (HID @ w2 + b2) * C ; agg[dst] += W * H[src]
    gemm_wmma<_Float16, true, false, EPI_SCATTER><<<N_EDGES / 32, 128, 0, stream>>>(
        HID, NF, NF, w2, b2, AGG, ei, Cbuf, Hbuf);

    // 6) T = ssp(agg @ lin2 + lin2_b)                     [N,256] (reuse Hbuf)
    gemm_wmma<float, true, true, EPI_STORE_F32><<<N_NODES / 32, 128, 0, stream>>>(
        AGG, NF, NF, lin2, lin2b, Hbuf, nullptr, nullptr, nullptr);

    // 7) out = T @ lin_w + lin_b                          [N,256]
    gemm_wmma<float, true, false, EPI_STORE_F32><<<N_NODES / 32, 128, 0, stream>>>(
        Hbuf, NF, NF, linw, linb, out, nullptr, nullptr, nullptr);
}
